// Attention_model_11931419148749
// MI455X (gfx1250) — matrix operand order repaired and checked by reading, not compiled, not run on hardware
//
#include <hip/hip_runtime.h>
#include <hip/hip_bf16.h>
#include <math.h>

#define B_ 64
#define L_ 256
#define D_ 1024
#define H_ 1024
#define U_ 1024
#define MT 32   // rows per workgroup in the fused scorer

typedef __attribute__((ext_vector_type(16))) __bf16 v16bf;
typedef __attribute__((ext_vector_type(4)))  __bf16 bf16x4;
typedef __attribute__((ext_vector_type(2)))  __bf16 bf16x2;
typedef __attribute__((ext_vector_type(8)))  float  v8f;

static __device__ __forceinline__ __bf16 f2bf(float x) { return (__bf16)x; }

static __device__ __forceinline__ void wait_async0() {
#if __has_builtin(__builtin_amdgcn_s_wait_asynccnt)
    __builtin_amdgcn_s_wait_asynccnt(0);
#else
    asm volatile("s_wait_asynccnt 0x0" ::: "memory");
#endif
}

// issue one 128-bit async global->LDS copy (ASYNCcnt tracked)
static __device__ __forceinline__ void async_copy_b128(unsigned lds_addr,
                                                       const void* gaddr) {
    asm volatile("global_load_async_to_lds_b128 %0, %1, off"
                 :: "v"(lds_addr), "v"((unsigned long long)(uintptr_t)gaddr)
                 : "memory");
}

// ---------------------------------------------------------------------------
// Kernel 0: W1T[n][k] = bf16(W1[k][n])  -- one-time transpose + convert
// grid (U/32, D/32), block 256
// ---------------------------------------------------------------------------
__global__ __launch_bounds__(256) void w1t_kernel(
    const float* __restrict__ W1w, __bf16* __restrict__ w1t)
{
    __shared__ __bf16 t[32 * 33];
    const int tid = threadIdx.x;
    const int n0 = blockIdx.x * 32;
    const int k0 = blockIdx.y * 32;
    {
        const int r  = tid >> 3;          // k-local
        const int c4 = (tid & 7) << 2;    // n-local
        const float4 v = *(const float4*)(W1w + (size_t)(k0 + r) * U_ + n0 + c4);
        __bf16* d = &t[r * 33 + c4];
        d[0] = f2bf(v.x); d[1] = f2bf(v.y); d[2] = f2bf(v.z); d[3] = f2bf(v.w);
    }
    __syncthreads();
    {
        const int rn = tid >> 3;          // n-local
        const int k4 = (tid & 7) << 2;    // k-local
        bf16x4 o;
        o[0] = t[(k4 + 0) * 33 + rn];
        o[1] = t[(k4 + 1) * 33 + rn];
        o[2] = t[(k4 + 2) * 33 + rn];
        o[3] = t[(k4 + 3) * 33 + rn];
        *(bf16x4*)(w1t + (size_t)(n0 + rn) * D_ + k0 + k4) = o;
    }
}

// ---------------------------------------------------------------------------
// Kernel 1: h_proj[b][u] = hidden[b] . W2[:,u] + W2_b[u]   (tiny, VALU)
// grid (U/256, B), block 256
// ---------------------------------------------------------------------------
__global__ __launch_bounds__(256) void hproj_kernel(
    const float* __restrict__ hidden, const float* __restrict__ W2w,
    const float* __restrict__ W2b, float* __restrict__ hproj)
{
    const int u = blockIdx.x * 256 + threadIdx.x;
    const int b = blockIdx.y;
    __shared__ float hrow[H_];
    for (int i = threadIdx.x; i < H_; i += 256) hrow[i] = hidden[b * H_ + i];
    __syncthreads();
    float acc = W2b[u];
#pragma unroll 8
    for (int h = 0; h < H_; ++h) acc = fmaf(hrow[h], W2w[(size_t)h * U_ + u], acc);
    hproj[b * U_ + u] = acc;
}

// ---------------------------------------------------------------------------
// Kernel 2: fused  logits[row] = sum_u tanh(feat@W1 + W1_b + hproj) * V_w[u]
// bf16 WMMA / f32 accum. 32 rows per workgroup, 8 waves, each wave owns
// 64 columns per half-pass (2 passes over U=1024). B tiles arrive via
// async global->LDS b128 copies from the pre-transposed bf16 W1T.
// ---------------------------------------------------------------------------
__global__ __launch_bounds__(256) void score_logits_kernel(
    const float* __restrict__ feat, const __bf16* __restrict__ w1t,
    const float* __restrict__ W1b, const float* __restrict__ Vw,
    const float* __restrict__ hproj, float* __restrict__ logits)
{
    __shared__ __bf16 lA[MT * 36];      // A tile 32x32, row stride 36 (pad)
    __shared__ __bf16 lB[512 * 40];     // B^T tile [n][k], stride 40 (16B align, pad)
    __shared__ float  red[MT];          // per-row logit accumulation

    const int tid  = threadIdx.x;
    const int wave = tid >> 5;
    const int lane = tid & 31;
    const int hi   = lane >> 4;         // half-wave select (ISA layout)
    const int lnib = lane & 15;
    const int row0 = blockIdx.x * MT;   // 32 | 256 so b is uniform per block
    const int b    = row0 / L_;
    const float* hp = hproj + (size_t)b * U_;

    if (tid < MT) red[tid] = 0.0f;

    for (int p = 0; p < 2; ++p) {       // two 512-column half-passes over U
        v8f acc[2][4] = {};
        for (int kt = 0; kt < D_ / 32; ++kt) {
            const int k0 = kt * 32;
            __syncthreads();            // previous readers done before restage

            // ---- stage B: 512 n-rows x 32 k bf16, pure async row copies --
#pragma unroll
            for (int i = 0; i < 8; ++i) {
                const int ch = i * 256 + tid;        // 0..2047 16B chunks
                const int n  = ch >> 2;              // 0..511
                const int cq = ch & 3;               // chunk within 64B row
                const __bf16* gsrc =
                    w1t + (size_t)(p * 512 + n) * D_ + k0 + cq * 8;
                const unsigned la = (unsigned)(uintptr_t)&lB[n * 40 + cq * 8];
                async_copy_b128(la, gsrc);
            }
            // ---- stage A: 32 rows x 32 k, fp32 -> bf16 -------------------
            {
                const int ar = tid >> 3;
                const int ac = (tid & 7) << 2;
                const float* gsrc = feat + (size_t)(row0 + ar) * D_ + k0 + ac;
                const float4 v = *(const float4*)gsrc;
                bf16x4 o;
                o[0] = f2bf(v.x); o[1] = f2bf(v.y);
                o[2] = f2bf(v.z); o[3] = f2bf(v.w);
                *(bf16x4*)&lA[ar * 36 + ac] = o;
                if (kt + 1 < D_ / 32)                // hint next A tile
                    __builtin_prefetch(gsrc + 32, 0, 1);
            }
            wait_async0();              // ASYNCcnt not covered by barrier
            __syncthreads();

            // ---- A fragments (ISA 16-bit A 16x32 layout) -----------------
            v16bf afrag[2];
#pragma unroll
            for (int s = 0; s < 2; ++s) {
                const int m = s * 16 + lnib;
#pragma unroll
                for (int j = 0; j < 8; ++j) {
                    const int k = 2 * j + 8 * hi + ((j >= 4) ? 8 : 0);
                    const bf16x2 pr = *(const bf16x2*)&lA[m * 36 + k];
                    afrag[s][2 * j]     = pr.x;
                    afrag[s][2 * j + 1] = pr.y;
                }
            }
            // ---- B fragments + WMMA --------------------------------------
#pragma unroll
            for (int nt = 0; nt < 4; ++nt) {
                const int nloc = wave * 64 + nt * 16 + lnib;
                v16bf bfrag;
#pragma unroll
                for (int j = 0; j < 8; ++j) {
                    const bf16x2 pr = *(const bf16x2*)&lB[nloc * 40 + 8 * hi + 2 * j + ((j >= 4) ? 8 : 0)];
                    bfrag[2 * j]     = pr.x;
                    bfrag[2 * j + 1] = pr.y;
                }
                acc[0][nt] = __builtin_amdgcn_wmma_f32_16x16x32_bf16(
                    false, afrag[0], false, bfrag, (short)0, acc[0][nt], false, false);
                acc[1][nt] = __builtin_amdgcn_wmma_f32_16x16x32_bf16(
                    false, afrag[1], false, bfrag, (short)0, acc[1][nt], false, false);
            }
        }

        // ---- epilogue: tanh, dot with V_w, per-row reduction -------------
        float rowsum[16];
#pragma unroll
        for (int i = 0; i < 16; ++i) rowsum[i] = 0.0f;
#pragma unroll
        for (int nt = 0; nt < 4; ++nt) {
            const int col    = p * 512 + wave * 64 + nt * 16 + lnib;
            const float bias = W1b[col] + hp[col];
            const float wv   = Vw[col];
#pragma unroll
            for (int s = 0; s < 2; ++s)
#pragma unroll
                for (int r = 0; r < 8; ++r)
                    rowsum[s * 8 + r] += tanhf(acc[s][nt][r] + bias) * wv;
        }
#pragma unroll
        for (int i = 0; i < 16; ++i) {
            const int rl = (i >> 3) * 16 + 8 * hi + (i & 7);   // D-tile row map
            atomicAdd(&red[rl], rowsum[i]);
        }
    }
    __syncthreads();
    if (tid < MT) logits[row0 + tid] = red[tid];
}

// ---------------------------------------------------------------------------
// Kernel 3: softmax over L, then context[b] = sum_l w[l] * features[b][l][:]
// grid B, block L_ (=256)
// ---------------------------------------------------------------------------
__global__ __launch_bounds__(256) void softmax_ctx_kernel(
    const float* __restrict__ feat, const float* __restrict__ logits,
    const float* __restrict__ Vb, float* __restrict__ out_ctx,
    float* __restrict__ out_w)
{
    __shared__ float sdata[L_];
    __shared__ float wbuf[L_];
    const int b = blockIdx.x;
    const int tid = threadIdx.x;

    const float lg = logits[b * L_ + tid] + Vb[0];
    sdata[tid] = lg;
    __syncthreads();
    for (int s = 128; s > 0; s >>= 1) {
        if (tid < s) sdata[tid] = fmaxf(sdata[tid], sdata[tid + s]);
        __syncthreads();
    }
    const float mx = sdata[0];
    __syncthreads();
    const float e = __expf(lg - mx);
    sdata[tid] = e;
    __syncthreads();
    for (int s = 128; s > 0; s >>= 1) {
        if (tid < s) sdata[tid] += sdata[tid + s];
        __syncthreads();
    }
    const float w = e / sdata[0];
    wbuf[tid] = w;
    out_w[b * L_ + tid] = w;
    __syncthreads();

    const int d0 = tid * 4;
    float ax = 0.f, ay = 0.f, az = 0.f, aw = 0.f;
    for (int l = 0; l < L_; ++l) {
        const float wl = wbuf[l];
        const float4 f = *(const float4*)(feat + ((size_t)(b * L_ + l)) * D_ + d0);
        ax = fmaf(wl, f.x, ax); ay = fmaf(wl, f.y, ay);
        az = fmaf(wl, f.z, az); aw = fmaf(wl, f.w, aw);
    }
    float4 o; o.x = ax; o.y = ay; o.z = az; o.w = aw;
    *(float4*)(out_ctx + (size_t)b * D_ + d0) = o;
}

// ---------------------------------------------------------------------------
extern "C" void kernel_launch(void* const* d_in, const int* in_sizes, int n_in,
                              void* d_out, int out_size, void* d_ws, size_t ws_size,
                              hipStream_t stream)
{
    const float* features = (const float*)d_in[0];   // [B, L, D]
    const float* hidden   = (const float*)d_in[1];   // [B, H]
    const float* W1w      = (const float*)d_in[2];   // [D, U]
    const float* W1b      = (const float*)d_in[3];   // [U]
    const float* W2w      = (const float*)d_in[4];   // [H, U]
    const float* W2b      = (const float*)d_in[5];   // [U]
    const float* Vw       = (const float*)d_in[6];   // [U, 1]
    const float* Vb       = (const float*)d_in[7];   // [1]

    float* out_ctx = (float*)d_out;                  // [B, D]
    float* out_w   = out_ctx + (size_t)B_ * D_;      // [B, L, 1]

    float*  hproj  = (float*)d_ws;                   // [B, U]    256 KB
    float*  logits = hproj + (size_t)B_ * U_;        // [B, L]     64 KB
    __bf16* w1t    = (__bf16*)(logits + (size_t)B_ * L_);  // [U, D] bf16, 2 MB

    w1t_kernel<<<dim3(U_ / 32, D_ / 32), 256, 0, stream>>>(W1w, w1t);
    hproj_kernel<<<dim3(U_ / 256, B_), 256, 0, stream>>>(hidden, W2w, W2b, hproj);
    score_logits_kernel<<<(B_ * L_) / MT, 256, 0, stream>>>(features, w1t, W1b, Vw, hproj, logits);
    softmax_ctx_kernel<<<B_, L_, 0, stream>>>(features, logits, Vb, out_ctx, out_w);
}
